// KineTransition_30296699306309
// MI455X (gfx1250) — matrix operand order repaired and checked by reading, not compiled, not run on hardware
//
#include <hip/hip_runtime.h>
#include <math.h>

// ---- WMMA vector types (gfx1250, wave32) ----
typedef __attribute__((ext_vector_type(16))) _Float16 v16h;
typedef __attribute__((ext_vector_type(8)))  _Float16 v8h;
typedef __attribute__((ext_vector_type(8)))  float    v8f;

#define VNUM   32
#define FNUM   7
#define SD     224          // V*F state dim
#define ADIM   2
#define HID    256
#define KPAD   256          // padded K for all layers
#define NOUT   64           // (V-1)*2 = 62 padded to 64
#define TM     64           // batch rows per workgroup
#define NTHR   256          // 8 wave32s
#define DT_    0.1f
#define MAXACC 5.0f
#define MAXSTR 0.6108652381980153f
#define HDG_K  (0.1f / 3.0f)   // DT / WHEELBASE

// Fast activations: single v_exp_f32 + single v_rcp_f32 each
__device__ __forceinline__ float fast_sigmoid(float x) {
    return __builtin_amdgcn_rcpf(1.0f + __expf(-x));
}
__device__ __forceinline__ float fast_tanh(float x) {
    // tanh(x) = 1 - 2/(e^{2x}+1)
    return 1.0f - 2.0f * __builtin_amdgcn_rcpf(__expf(2.0f * x) + 1.0f);
}

// ---------------------------------------------------------------------------
// Weight prep: fp32 -> f16, with K padding (226->256) and N padding (62->64).
// W1h [256][256], W2h [256][256], W3h [64][256]  (row-major, K contiguous)
// ---------------------------------------------------------------------------
__global__ __launch_bounds__(NTHR)
void prep_weights(const float* __restrict__ W1, const float* __restrict__ W2,
                  const float* __restrict__ W3,
                  _Float16* __restrict__ W1h, _Float16* __restrict__ W2h,
                  _Float16* __restrict__ W3h)
{
    int i = blockIdx.x * NTHR + threadIdx.x;
    if (i < 256 * 256) {
        int n = i >> 8, k = i & 255;
        W1h[i] = (_Float16)((k < SD + ADIM) ? W1[n * (SD + ADIM) + k] : 0.0f);
    } else if (i < 2 * 256 * 256) {
        int j = i - 256 * 256;
        W2h[j] = (_Float16)W2[j];
    } else if (i < 2 * 256 * 256 + NOUT * 256) {
        int j = i - 2 * 256 * 256;
        int n = j >> 8;
        W3h[j] = (_Float16)((n < (VNUM - 1) * 2) ? W3[j] : 0.0f);
    }
}

// ---------------------------------------------------------------------------
// One hidden GEMM layer: out[TM][256] = act(in[TM][256] @ Wh^T + bias), f16 out
// Layouts used here (A/B: this code's contiguous K order, same for both; C/D per ISA):
//   A 16x32 f16: lane L holds row (L&15); 16 contiguous K at (L>>4)*16
//                -> two contiguous v8h LDS loads
//   B 32x16 f16: lane L holds col (L&15); 16 contiguous K at (L>>4)*16
//   C/D 16x16 f32: VGPR i -> row i + (L>>4)*8, col (L&15)
// ---------------------------------------------------------------------------
__device__ __forceinline__ void gemm_hidden(
    const _Float16 (*in)[KPAD], const _Float16* __restrict__ Wh,
    const float* __restrict__ bias, _Float16 (*outp)[KPAD],
    int lane, int wv)
{
    const int lm = lane & 15;
    const int kh = lane >> 4;          // 0 or 1
    // 4 M-tiles x 16 N-tiles = 64 tiles over 8 waves
    for (int t = wv; t < 64; t += 8) {
        const int mt = t >> 4, nt = t & 15;
        const int n  = nt * 16 + lm;
        const int ar = mt * 16 + lm;
        const float bv = bias[n];
        v8f acc = { bv, bv, bv, bv, bv, bv, bv, bv };
        const _Float16* wrow = Wh + (size_t)n * KPAD + kh * 16;
        const _Float16* arow = &in[ar][kh * 16];
#pragma unroll
        for (int k0 = 0; k0 < KPAD; k0 += 32) {
            union { v16h v; v8h h[2]; } a;
            a.h[0] = *(const v8h*)(arow + k0);
            a.h[1] = *(const v8h*)(arow + k0 + 8);
            v16h b = *(const v16h*)(wrow + k0);
            acc = __builtin_amdgcn_wmma_f32_16x16x32_f16(
                false, a.v, false, b, (short)0, acc, false, false);
        }
        const int rb = mt * 16 + kh * 8;
#pragma unroll
        for (int i = 0; i < 8; i++)
            outp[rb + i][n] = (_Float16)fast_sigmoid(acc[i]);
    }
}

// Output layer: r[TM][64] = tanh(in @ W3h^T + b3), fp32 result kept in LDS
__device__ __forceinline__ void gemm_ctrl(
    const _Float16 (*in)[KPAD], const _Float16* __restrict__ Wh,
    const float* __restrict__ bias, float (*rout)[NOUT],
    int lane, int wv)
{
    const int lm = lane & 15;
    const int kh = lane >> 4;
    // 4 M-tiles x 4 N-tiles = 16 tiles over 8 waves
    for (int t = wv; t < 16; t += 8) {
        const int mt = t >> 2, nt = t & 3;
        const int n  = nt * 16 + lm;
        const int ar = mt * 16 + lm;
        const float bv = (n < (VNUM - 1) * 2) ? bias[n] : 0.0f;
        v8f acc = { bv, bv, bv, bv, bv, bv, bv, bv };
        const _Float16* wrow = Wh + (size_t)n * KPAD + kh * 16;
        const _Float16* arow = &in[ar][kh * 16];
#pragma unroll
        for (int k0 = 0; k0 < KPAD; k0 += 32) {
            union { v16h v; v8h h[2]; } a;
            a.h[0] = *(const v8h*)(arow + k0);
            a.h[1] = *(const v8h*)(arow + k0 + 8);
            v16h b = *(const v16h*)(wrow + k0);
            acc = __builtin_amdgcn_wmma_f32_16x16x32_f16(
                false, a.v, false, b, (short)0, acc, false, false);
        }
        const int rb = mt * 16 + kh * 8;
#pragma unroll
        for (int i = 0; i < 8; i++)
            rout[rb + i][n] = fast_tanh(acc[i]);
    }
}

// ---------------------------------------------------------------------------
// Fused kernel: load tile -> 3 WMMA GEMM layers -> bicycle kinematics epilogue
// ---------------------------------------------------------------------------
__global__ __launch_bounds__(NTHR)
void kine_fused(const float* __restrict__ state,
                const float* __restrict__ action,
                const _Float16* __restrict__ W1h, const float* __restrict__ b1,
                const _Float16* __restrict__ W2h, const float* __restrict__ b2,
                const _Float16* __restrict__ W3h, const float* __restrict__ b3,
                float* __restrict__ out, int batch)
{
    __shared__ _Float16 lA[TM][KPAD];   // 32 KB
    __shared__ _Float16 lB[TM][KPAD];   // 32 KB
    __shared__ float    lR[TM][NOUT];   // 16 KB

    const int tid  = threadIdx.x;
    const int lane = tid & 31;
    const int wv   = tid >> 5;
    const int row0 = blockIdx.x * TM;

    // Phase 0: build X = concat(state, action) as f16, zero-padded to K=256
    for (int e = tid; e < TM * KPAD; e += NTHR) {
        const int r = e >> 8, k = e & 255;
        float v = 0.0f;
        if (row0 + r < batch) {
            if (k < SD)             v = state[(size_t)(row0 + r) * SD + k];
            else if (k < SD + ADIM) v = action[(size_t)(row0 + r) * ADIM + (k - SD)];
        }
        lA[r][k] = (_Float16)v;
    }
    __syncthreads();

    gemm_hidden(lA, W1h, b1, lB, lane, wv);   // h1 = sigmoid(x W1^T + b1)
    __syncthreads();
    gemm_hidden(lB, W2h, b2, lA, lane, wv);   // h2 = sigmoid(h1 W2^T + b2)
    __syncthreads();
    gemm_ctrl(lA, W3h, b3, lR, lane, wv);     // r  = tanh(h2 W3^T + b3)
    __syncthreads();

    // Phase 4: masked bicycle-kinematics update, write all 7 features
    for (int p = tid; p < TM * VNUM; p += NTHR) {
        const int r   = p >> 5;
        const int veh = p & 31;
        const int gr  = row0 + r;
        if (gr >= batch) break;
        const size_t base = (size_t)gr * SD + (size_t)veh * FNUM;

        const float presence = state[base + 0];
        const float px  = state[base + 1];
        const float py  = state[base + 2];
        const float f3  = state[base + 3];
        const float f4  = state[base + 4];
        const float hdg = state[base + 5];
        const float spd = state[base + 6];

        float ac, st;
        if (veh == 0) {
            ac = action[(size_t)gr * ADIM + 0];
            st = action[(size_t)gr * ADIM + 1];
        } else {
            ac = lR[r][(veh - 1) * 2 + 0] * MAXACC;
            st = lR[r][(veh - 1) * 2 + 1] * MAXSTR;
        }

        const float nx = px  + spd * __cosf(hdg) * DT_;
        const float ny = py  + spd * __sinf(hdg) * DT_;
        const float nh = hdg + spd * __tanf(st) * HDG_K;
        const float ns = spd + ac * DT_;
        const bool  m  = (presence != 0.0f);

        out[base + 0] = presence;
        out[base + 1] = m ? nx : px;
        out[base + 2] = m ? ny : py;
        out[base + 3] = f3;
        out[base + 4] = f4;
        out[base + 5] = m ? nh : hdg;
        out[base + 6] = m ? ns : spd;
    }
}

// ---------------------------------------------------------------------------
extern "C" void kernel_launch(void* const* d_in, const int* in_sizes, int n_in,
                              void* d_out, int out_size, void* d_ws, size_t ws_size,
                              hipStream_t stream)
{
    const float* state  = (const float*)d_in[0];
    const float* action = (const float*)d_in[1];
    const float* W1 = (const float*)d_in[2];
    const float* b1 = (const float*)d_in[3];
    const float* W2 = (const float*)d_in[4];
    const float* b2 = (const float*)d_in[5];
    const float* W3 = (const float*)d_in[6];
    const float* b3 = (const float*)d_in[7];
    float* out = (float*)d_out;

    const int batch = in_sizes[0] / SD;

    // workspace layout: f16 weights (K padded to 256)
    _Float16* W1h = (_Float16*)d_ws;            // 256*256
    _Float16* W2h = W1h + 256 * 256;            // 256*256
    _Float16* W3h = W2h + 256 * 256;            // 64*256

    const int prep_elems = 2 * 256 * 256 + NOUT * 256;
    prep_weights<<<(prep_elems + NTHR - 1) / NTHR, NTHR, 0, stream>>>(
        W1, W2, W3, W1h, W2h, W3h);

    const int nblocks = (batch + TM - 1) / TM;
    kine_fused<<<nblocks, NTHR, 0, stream>>>(
        state, action, W1h, b1, W2h, b2, W3h, b3, out, batch);
}
